// HybridViT_Augmented_40699110097348
// MI455X (gfx1250) — matrix operand order repaired and checked by reading, not compiled, not run on hardware
//
#include <hip/hip_runtime.h>
#include <math.h>

// ---------------- problem constants ----------------
#define Bsz   1024
#define Cch   3
#define Himg  32
#define Wimg  32
#define Pp    4
#define Dm    128
#define NHn   4
#define DHn   32
#define FFn   256
#define Ln    4
#define Tn    65                 // 8*8 patches + cls
#define MROWS (Bsz * Tn)         // 66560, multiple of 32
#define NQn   4

typedef __attribute__((ext_vector_type(16))) __bf16 v16bf;
typedef __attribute__((ext_vector_type(8)))  float  v8f;

// ---------------- helpers ----------------
__device__ __forceinline__ unsigned short bf16_rne(float x) {
  union { float f; unsigned u; } a; a.f = x;
  unsigned ua = a.u + (0x7FFFu + ((a.u >> 16) & 1u));
  return (unsigned short)(ua >> 16);
}

// A fragment: 16x32 bf16 (M x K), lane (r, r+16) holds row r.
// lanes 0-15 (half=0): VGPR v<4 -> K=2v,2v+1 ; v>=4 -> K=2v+8,2v+9
// lanes 16-31 (half=1): same + 8
// => per lane: two contiguous 16B runs at K = 8*half and K = 16 + 8*half
__device__ __forceinline__ v16bf load_a_frag(const unsigned short* __restrict__ A,
                                             int row0, int kb, int ldk,
                                             int half, int r) {
  const unsigned short* p = A + (size_t)(row0 + r) * ldk + kb * 32 + 8 * half;
  union { uint4 q[2]; v16bf v; } f;
  f.q[0] = *(const uint4*)(p);
  f.q[1] = *(const uint4*)(p + 16);
  return f.v;
}

// B fragment: 32x16 bf16 (K x N), lane n holds column n.
// per lane: two contiguous 16B runs at K = 8*half and K = 16 + 8*half (same as A)
// Weights stored row-major [N, K] (reference w[out,in]), bf16.
__device__ __forceinline__ v16bf load_b_frag(const unsigned short* __restrict__ W,
                                             int n0, int kb, int ldk,
                                             int half, int r) {
  const unsigned short* p = W + (size_t)(n0 + r) * ldk + kb * 32 + 8 * half;
  union { uint4 q[2]; v16bf v; } f;
  f.q[0] = *(const uint4*)(p);
  f.q[1] = *(const uint4*)(p + 16);
  return f.v;
}

// ---------------- WMMA GEMM:  Y[M,N] = A[M,K] * W[N,K]^T (+bias)(+relu)(+resid) ----------------
// grid.x = M/32 tiles (MPER=2); block = 256 threads (8 waves); wave w owns N-tiles w, w+8, ...
// Output: fp32 Y (nullable) and/or bf16 Ybf (nullable).
template <int NPER>
__global__ void gemm_bf16_wmma(const unsigned short* __restrict__ A,
                               const unsigned short* __restrict__ W,
                               const float* __restrict__ bias,
                               const float* __restrict__ resid,
                               float* __restrict__ Y,
                               unsigned short* __restrict__ Ybf,
                               int K, int N, int relu) {
  const int wave = threadIdx.x >> 5;
  const int lane = threadIdx.x & 31;
  const int half = lane >> 4;
  const int r    = lane & 15;
  const int row0 = blockIdx.x * 32;

  v8f acc[2][NPER];
#pragma unroll
  for (int mi = 0; mi < 2; ++mi)
#pragma unroll
    for (int j = 0; j < NPER; ++j)
#pragma unroll
      for (int e = 0; e < 8; ++e) acc[mi][j][e] = 0.0f;

  const int kblocks = K >> 5;
  for (int kb = 0; kb < kblocks; ++kb) {
    v16bf a0 = load_a_frag(A, row0,      kb, K, half, r);
    v16bf a1 = load_a_frag(A, row0 + 16, kb, K, half, r);
#pragma unroll
    for (int j = 0; j < NPER; ++j) {
      int n0 = (j * 8 + wave) * 16;
      v16bf bfrag = load_b_frag(W, n0, kb, K, half, r);
      acc[0][j] = __builtin_amdgcn_wmma_f32_16x16x32_bf16(
          false, a0, false, bfrag, (short)0, acc[0][j], false, false);
      acc[1][j] = __builtin_amdgcn_wmma_f32_16x16x32_bf16(
          false, a1, false, bfrag, (short)0, acc[1][j], false, false);
    }
  }

  // C/D layout: VGPR v -> row (v + 8*half), col r within each 16x16 tile
#pragma unroll
  for (int mi = 0; mi < 2; ++mi) {
#pragma unroll
    for (int j = 0; j < NPER; ++j) {
      int col = (j * 8 + wave) * 16 + r;
      float bval = bias ? bias[col] : 0.0f;
#pragma unroll
      for (int v = 0; v < 8; ++v) {
        int m = row0 + mi * 16 + v + 8 * half;
        float val = acc[mi][j][v] + bval;
        if (relu) val = fmaxf(val, 0.0f);
        size_t off = (size_t)m * N + col;
        if (resid) val += resid[off];
        if (Y)   Y[off]   = val;
        if (Ybf) Ybf[off] = bf16_rne(val);
      }
    }
  }
}

// ---------------- fp32 -> bf16 weight conversion ----------------
__global__ void f32_to_bf16_kernel(const float* __restrict__ src,
                                   unsigned short* __restrict__ dst, int n) {
  int i = blockIdx.x * blockDim.x + threadIdx.x;
  if (i < n) dst[i] = bf16_rne(src[i]);
}

// ---------------- patch embedding + pos embed ----------------
__global__ void patch_embed_kernel(const float* __restrict__ x,
                                   const float* __restrict__ pw,
                                   const float* __restrict__ pb,
                                   const float* __restrict__ pos,
                                   float* __restrict__ Hm,
                                   unsigned short* __restrict__ Hbf) {
  int idx = blockIdx.x * blockDim.x + threadIdx.x;  // B*64*128 total
  int d  = idx & 127;
  int pj = (idx >> 7) & 63;
  int b  = idx >> 13;
  int i = pj >> 3, j = pj & 7;
  const float* xb = x + (size_t)b * Cch * Himg * Wimg;
  const float* wd = pw + (size_t)d * Cch * Pp * Pp;
  float acc = pb[d];
#pragma unroll
  for (int c = 0; c < Cch; ++c)
#pragma unroll
    for (int p = 0; p < Pp; ++p)
#pragma unroll
      for (int q = 0; q < Pp; ++q)
        acc += xb[c * 1024 + (i * 4 + p) * 32 + (j * 4 + q)] *
               wd[(c * Pp + p) * Pp + q];
  int t = 1 + pj;
  float val = acc + pos[t * Dm + d];
  size_t off = ((size_t)b * Tn + t) * Dm + d;
  Hm[off]  = val;
  Hbf[off] = bf16_rne(val);
}

__global__ void cls_pos_kernel(const float* __restrict__ cls,
                               const float* __restrict__ pos,
                               float* __restrict__ Hm,
                               unsigned short* __restrict__ Hbf) {
  int idx = blockIdx.x * blockDim.x + threadIdx.x;  // B*128 total
  int d = idx & 127, b = idx >> 7;
  float val = cls[d] + pos[d];
  size_t off = (size_t)b * Tn * Dm + d;
  Hm[off]  = val;
  Hbf[off] = bf16_rne(val);
}

// ---------------- attention: one block per (b, head); bf16 output ----------------
__global__ void attn_kernel(const float* __restrict__ QKV,        // [B*T, 384] fp32
                            unsigned short* __restrict__ Obf) {   // [B*T, 128] bf16
  const int bh = blockIdx.x;
  const int b = bh >> 2, h = bh & 3;
  __shared__ float Ks[Tn][DHn + 1];
  __shared__ float Vs[Tn][DHn + 1];
  const float* base = QKV + (size_t)b * Tn * 384;
  for (int idx = threadIdx.x; idx < Tn * DHn; idx += blockDim.x) {
    int t = idx >> 5, d = idx & 31;
    Ks[t][d] = base[(size_t)t * 384 + 128 + h * DHn + d];
    Vs[t][d] = base[(size_t)t * 384 + 256 + h * DHn + d];
  }
  __syncthreads();
  const int q = threadIdx.x;
  if (q < Tn) {
    const float scale = 0.17677669529663687f;  // 1/sqrt(32)
    float qv[DHn];
#pragma unroll
    for (int d = 0; d < DHn; ++d) qv[d] = base[(size_t)q * 384 + h * DHn + d];
    // pass 1: max logit
    float mx = -3.4e38f;
    for (int k = 0; k < Tn; ++k) {
      float s = 0.0f;
#pragma unroll
      for (int d = 0; d < DHn; ++d) s += qv[d] * Ks[k][d];
      mx = fmaxf(mx, s * scale);
    }
    // pass 2: exp-weighted sum of V
    float o[DHn];
#pragma unroll
    for (int d = 0; d < DHn; ++d) o[d] = 0.0f;
    float sum = 0.0f;
    for (int k = 0; k < Tn; ++k) {
      float s = 0.0f;
#pragma unroll
      for (int d = 0; d < DHn; ++d) s += qv[d] * Ks[k][d];
      float e = __expf(s * scale - mx);
      sum += e;
#pragma unroll
      for (int d = 0; d < DHn; ++d) o[d] += e * Vs[k][d];
    }
    float inv = 1.0f / sum;
    unsigned short* op = Obf + ((size_t)b * Tn + q) * Dm + h * DHn;
#pragma unroll
    for (int d = 0; d < DHn; ++d) op[d] = bf16_rne(o[d] * inv);
  }
}

// ---------------- layernorm: one wave per row of 128; fp32 + bf16 outputs ----------------
__global__ void ln_kernel(const float* __restrict__ Y,
                          const float* __restrict__ gamma,
                          const float* __restrict__ beta,
                          float* __restrict__ Hout,
                          unsigned short* __restrict__ Hbf) {
  const int wave = threadIdx.x >> 5;
  const int lane = threadIdx.x & 31;
  const size_t row = (size_t)blockIdx.x * (blockDim.x >> 5) + wave;
  if (row >= MROWS) return;
  const float* y = Y + row * Dm;
  float v0 = y[lane], v1 = y[lane + 32], v2 = y[lane + 64], v3 = y[lane + 96];
  float s = v0 + v1 + v2 + v3;
#pragma unroll
  for (int o = 16; o > 0; o >>= 1) s += __shfl_xor(s, o, 32);
  float mean = s * (1.0f / 128.0f);
  float d0 = v0 - mean, d1 = v1 - mean, d2 = v2 - mean, d3 = v3 - mean;
  float ss = d0 * d0 + d1 * d1 + d2 * d2 + d3 * d3;
#pragma unroll
  for (int o = 16; o > 0; o >>= 1) ss += __shfl_xor(ss, o, 32);
  float inv = rsqrtf(ss * (1.0f / 128.0f) + 1e-5f);
  float r0 = d0 * inv * gamma[lane]      + beta[lane];
  float r1 = d1 * inv * gamma[lane + 32] + beta[lane + 32];
  float r2 = d2 * inv * gamma[lane + 64] + beta[lane + 64];
  float r3 = d3 * inv * gamma[lane + 96] + beta[lane + 96];
  float* out = Hout + row * Dm;
  out[lane] = r0; out[lane + 32] = r1; out[lane + 64] = r2; out[lane + 96] = r3;
  unsigned short* ob = Hbf + row * Dm;
  ob[lane]      = bf16_rne(r0);
  ob[lane + 32] = bf16_rne(r1);
  ob[lane + 64] = bf16_rne(r2);
  ob[lane + 96] = bf16_rne(r3);
}

// ---------------- classifier head: one thread per batch row ----------------
__global__ void head_kernel(const float* __restrict__ Hm,
                            const float* __restrict__ qr1w, const float* __restrict__ qr1b,
                            const float* __restrict__ qr2w, const float* __restrict__ qr2b,
                            const float* __restrict__ qw,
                            const float* __restrict__ clfw, const float* __restrict__ clfb,
                            float* __restrict__ out) {
  int b = blockIdx.x * blockDim.x + threadIdx.x;
  if (b >= Bsz) return;
  const float* cls = Hm + (size_t)b * Tn * Dm;  // h[b, 0, :]
  float r1[32];
  for (int n = 0; n < 32; ++n) {
    float s = qr1b[n];
    for (int k = 0; k < Dm; ++k) s += cls[k] * qr1w[n * Dm + k];
    r1[n] = fmaxf(s, 0.0f);
  }
  float qo[NQn];
  float cum = 1.0f;
  for (int j = 0; j < NQn; ++j) {
    float s = qr2b[j];
    for (int k = 0; k < 32; ++k) s += r1[k] * qr2w[j * 32 + k];
    float qi = tanhf(s);
    cum *= __cosf(qi) * __cosf(qw[j]);
    qo[j] = cum;
  }
  for (int o = 0; o < 10; ++o) {
    float s = clfb[o];
    for (int k = 0; k < Dm; ++k) s += cls[k] * clfw[o * (Dm + NQn) + k];
    for (int j = 0; j < NQn; ++j) s += qo[j] * clfw[o * (Dm + NQn) + Dm + j];
    out[b * 10 + o] = s;
  }
}

// ---------------- launcher ----------------
extern "C" void kernel_launch(void* const* d_in, const int* in_sizes, int n_in,
                              void* d_out, int out_size, void* d_ws, size_t ws_size,
                              hipStream_t stream) {
  const float* x        = (const float*)d_in[0];
  const float* patch_w  = (const float*)d_in[1];
  const float* patch_b  = (const float*)d_in[2];
  const float* cls_tok  = (const float*)d_in[3];
  const float* pos_emb  = (const float*)d_in[4];
  const float* qkv_w    = (const float*)d_in[5];
  const float* qkv_b    = (const float*)d_in[6];
  const float* out_w    = (const float*)d_in[7];
  const float* out_b    = (const float*)d_in[8];
  const float* ln1_s    = (const float*)d_in[9];
  const float* ln1_b    = (const float*)d_in[10];
  const float* ln2_s    = (const float*)d_in[11];
  const float* ln2_b    = (const float*)d_in[12];
  const float* ff1_w    = (const float*)d_in[13];
  const float* ff1_b    = (const float*)d_in[14];
  const float* ff2_w    = (const float*)d_in[15];
  const float* ff2_b    = (const float*)d_in[16];
  const float* qr1_w    = (const float*)d_in[17];
  const float* qr1_b    = (const float*)d_in[18];
  const float* qr2_w    = (const float*)d_in[19];
  const float* qr2_b    = (const float*)d_in[20];
  const float* q_wts    = (const float*)d_in[21];
  const float* clf_w    = (const float*)d_in[22];
  const float* clf_b    = (const float*)d_in[23];
  float* outp = (float*)d_out;

  // workspace layout (all sizes 256B-aligned by construction)
  char* p = (char*)d_ws;
  float* h    = (float*)p;  p += (size_t)MROWS * Dm  * sizeof(float);          // 34 MB  fp32 h
  float* bufA = (float*)p;  p += (size_t)MROWS * 384 * sizeof(float);          // 102 MB QKV fp32
  float* bufC = (float*)p;  p += (size_t)MROWS * Dm  * sizeof(float);          // 34 MB  pre-LN fp32
  unsigned short* h_bf = (unsigned short*)p; p += (size_t)MROWS * Dm  * 2;     // 17 MB  bf16 h
  unsigned short* O_bf = (unsigned short*)p; p += (size_t)MROWS * Dm  * 2;     // 17 MB  bf16 attn out
  unsigned short* F_bf = (unsigned short*)p; p += (size_t)MROWS * FFn * 2;     // 34 MB  bf16 FF1 out
  unsigned short* qkv_bf = (unsigned short*)p; p += (size_t)Ln * 384 * Dm  * 2;
  unsigned short* out_bf = (unsigned short*)p; p += (size_t)Ln * Dm  * Dm  * 2;
  unsigned short* ff1_bf = (unsigned short*)p; p += (size_t)Ln * FFn * Dm  * 2;
  unsigned short* ff2_bf = (unsigned short*)p; p += (size_t)Ln * Dm  * FFn * 2;

  // 1) weights -> bf16
  {
    int n1 = Ln * 384 * Dm, n2 = Ln * Dm * Dm, n3 = Ln * FFn * Dm, n4 = Ln * Dm * FFn;
    f32_to_bf16_kernel<<<(n1 + 255) / 256, 256, 0, stream>>>(qkv_w, qkv_bf, n1);
    f32_to_bf16_kernel<<<(n2 + 255) / 256, 256, 0, stream>>>(out_w, out_bf, n2);
    f32_to_bf16_kernel<<<(n3 + 255) / 256, 256, 0, stream>>>(ff1_w, ff1_bf, n3);
    f32_to_bf16_kernel<<<(n4 + 255) / 256, 256, 0, stream>>>(ff2_w, ff2_bf, n4);
  }

  // 2) patch embed + cls/pos (fp32 h + bf16 mirror)
  patch_embed_kernel<<<(Bsz * 64 * Dm) / 256, 256, 0, stream>>>(x, patch_w, patch_b, pos_emb, h, h_bf);
  cls_pos_kernel<<<(Bsz * Dm) / 256, 256, 0, stream>>>(cls_tok, pos_emb, h, h_bf);

  const int mtiles = MROWS / 32;  // 2080 (MPER=2)
  // 3) transformer layers
  for (int i = 0; i < Ln; ++i) {
    // QKV: [M,128] x [384,128]^T -> bufA [M,384] fp32
    gemm_bf16_wmma<3><<<mtiles, 256, 0, stream>>>(
        h_bf, qkv_bf + (size_t)i * 384 * Dm, qkv_b + (size_t)i * 384,
        nullptr, bufA, nullptr, Dm, 384, 0);
    // attention per (b, head) -> bf16 O
    attn_kernel<<<Bsz * NHn, 128, 0, stream>>>(bufA, O_bf);
    // out-proj + residual(h): [M,128] x [128,128]^T -> bufC fp32
    gemm_bf16_wmma<1><<<mtiles, 256, 0, stream>>>(
        O_bf, out_bf + (size_t)i * Dm * Dm, out_b + (size_t)i * Dm,
        h, bufC, nullptr, Dm, Dm, 0);
    // LN1 -> h (fp32) + h_bf
    ln_kernel<<<MROWS / 8, 256, 0, stream>>>(bufC, ln1_s + (size_t)i * Dm, ln1_b + (size_t)i * Dm, h, h_bf);
    // FF1 + relu: [M,128] x [256,128]^T -> F_bf (bf16 only)
    gemm_bf16_wmma<2><<<mtiles, 256, 0, stream>>>(
        h_bf, ff1_bf + (size_t)i * FFn * Dm, ff1_b + (size_t)i * FFn,
        nullptr, nullptr, F_bf, Dm, FFn, 1);
    // FF2 + residual(h): [M,256] x [128,256]^T -> bufC fp32
    gemm_bf16_wmma<1><<<mtiles, 256, 0, stream>>>(
        F_bf, ff2_bf + (size_t)i * Dm * FFn, ff2_b + (size_t)i * Dm,
        h, bufC, nullptr, FFn, Dm, 0);
    // LN2 -> h (fp32) + h_bf
    ln_kernel<<<MROWS / 8, 256, 0, stream>>>(bufC, ln2_s + (size_t)i * Dm, ln2_b + (size_t)i * Dm, h, h_bf);
  }

  // 4) head
  head_kernel<<<(Bsz + 255) / 256, 256, 0, stream>>>(
      h, qr1_w, qr1_b, qr2_w, qr2_b, q_wts, clf_w, clf_b, outp);
}
